// my_AugAttentionModule_4105988735249
// MI455X (gfx1250) — matrix operand order repaired and checked by reading, not compiled, not run on hardware
//
#include <hip/hip_runtime.h>
#include <math.h>

// ---------------------------------------------------------------------------
// AugAttention (VSS + rank-masked attention) for MI455X / gfx1250.
// All GEMMs through v_wmma_f32_16x16x32_f16 (wave32). Each wave computes a
// 32x32 output block (2x2 register tiling, 4 accumulators) so each K-step
// issues 4 WMMAs reusing 2 A-frags + 2 B-frags (~8 FLOP/byte from L2).
// Scan / LN / softmax / rank-mask in fp32 VALU. Scratch (<=122MB) L2-resident.
// ---------------------------------------------------------------------------

typedef __attribute__((ext_vector_type(16))) _Float16 v16h;
typedef __attribute__((ext_vector_type(8)))  float    v8f;

#define BB   2
#define CC   512
#define LL   1024
#define DD   1024
#define NST_ 16
#define RNK_ 32

// ---- fragment loaders -----------------------------------------------------
// A-frag element e: K = k0 + 16*kh + e  (16 contiguous K per lane)
__device__ __forceinline__ v16h load_a_frag(const float* __restrict__ rowp,
                                            int k0, int kh)
{
    const float* ap = rowp + k0 + 16 * kh;
    float4 f0 = *(const float4*)(ap);
    float4 f1 = *(const float4*)(ap + 4);
    float4 f2 = *(const float4*)(ap + 8);
    float4 f3 = *(const float4*)(ap + 12);
    v16h a;
    a[0]=(_Float16)f0.x; a[1]=(_Float16)f0.y; a[2]=(_Float16)f0.z; a[3]=(_Float16)f0.w;
    a[4]=(_Float16)f1.x; a[5]=(_Float16)f1.y; a[6]=(_Float16)f1.z; a[7]=(_Float16)f1.w;
    a[8]=(_Float16)f2.x; a[9]=(_Float16)f2.y; a[10]=(_Float16)f2.z; a[11]=(_Float16)f2.w;
    a[12]=(_Float16)f3.x; a[13]=(_Float16)f3.y; a[14]=(_Float16)f3.z; a[15]=(_Float16)f3.w;
    return a;
}

// B-frag element e: K = k0 + 16*kh + e  (16 contiguous K per lane)
__device__ __forceinline__ v16h load_bt_frag(const float* __restrict__ rowp,
                                             int k0, int kh)
{
    const float* bp = rowp + k0 + 16 * kh;
    float4 g0 = *(const float4*)(bp);
    float4 g1 = *(const float4*)(bp + 4);
    float4 g2 = *(const float4*)(bp + 8);
    float4 g3 = *(const float4*)(bp + 12);
    v16h b;
    b[0]=(_Float16)g0.x; b[1]=(_Float16)g0.y; b[2]=(_Float16)g0.z; b[3]=(_Float16)g0.w;
    b[4]=(_Float16)g1.x; b[5]=(_Float16)g1.y; b[6]=(_Float16)g1.z; b[7]=(_Float16)g1.w;
    b[8]=(_Float16)g2.x; b[9]=(_Float16)g2.y; b[10]=(_Float16)g2.z; b[11]=(_Float16)g2.w;
    b[12]=(_Float16)g3.x; b[13]=(_Float16)g3.y; b[14]=(_Float16)g3.z; b[15]=(_Float16)g3.w;
    return b;
}

__device__ __forceinline__ v16h load_bn_frag(const float* __restrict__ Bb,
                                             int ldb, int k0, int kh, int n)
{
    const float* bp = Bb + (size_t)(k0 + 16 * kh) * ldb + n;
    v16h b;
    #pragma unroll
    for (int e = 0; e < 16; ++e) b[e] = (_Float16)bp[(size_t)e * ldb];
    return b;
}

// ---------------------------------------------------------------------------
// Generic batched WMMA GEMM:  C = act(alpha * A[MxK] * B + bias + Res)
//  Wave computes a 32x32 block (2x2 tiles); block = 4 waves = 32x128.
//  grid = (N/128, M/32, batch)
//  A : row-major MxK (lda), per-batch offset (bz % aMod)*sA
//  B : BT=1 -> row-major NxK (ldb) ; BT=0 -> row-major KxN (ldb)
//  C : CT=0 -> C[m*ldc+n],  CT=1 -> C[n*ldc+m]
//  Res: optional residual, resT=1 -> Res[n*ldr+m]
//  ACT: 0=none, 1=relu, 2=softplus
// ---------------------------------------------------------------------------
template<bool BT, bool CT, int ACT>
__global__ __launch_bounds__(128)
void gemm_wmma(const float* __restrict__ A, int lda, long sA, int aMod,
               const float* __restrict__ Bm, int ldb, long sB,
               float* __restrict__ Cm, int ldc, long sC,
               const float* __restrict__ bias, int biasStride,
               const float* __restrict__ Res, int ldr, long sR, int resT,
               float alpha, int M, int N, int K)
{
    const int bz   = blockIdx.z;
    const float* Ab = A  + (long)(bz % aMod) * sA;
    const float* Bb = Bm + (long)bz * sB;
    float*       Cb = Cm + (long)bz * sC;

    const int lane = threadIdx.x & 31;
    const int wave = threadIdx.x >> 5;
    const int n0 = (blockIdx.x * 4 + wave) * 32;
    const int m0 = blockIdx.y * 32;
    if (n0 >= N || m0 >= M) return;   // wave-uniform: EXEC stays all-ones

    const int l15 = lane & 15;
    const int kh  = lane >> 4;

    const float* rowA0 = Ab + (size_t)(m0 + l15) * lda;
    const float* rowA1 = rowA0 + (size_t)16 * lda;
    const float* rowB0 = BT ? Bb + (size_t)(n0 + l15) * ldb : nullptr;
    const float* rowB1 = BT ? rowB0 + (size_t)16 * ldb : nullptr;

    v8f acc[2][2] = {};
    for (int k0 = 0; k0 < K; k0 += 32) {
        v16h a0 = load_a_frag(rowA0, k0, kh);
        v16h a1 = load_a_frag(rowA1, k0, kh);
        v16h b0, b1;
        if (BT) {
            b0 = load_bt_frag(rowB0, k0, kh);
            b1 = load_bt_frag(rowB1, k0, kh);
        } else {
            b0 = load_bn_frag(Bb, ldb, k0, kh, n0 + l15);
            b1 = load_bn_frag(Bb, ldb, k0, kh, n0 + 16 + l15);
        }
        acc[0][0] = __builtin_amdgcn_wmma_f32_16x16x32_f16(false, a0, false, b0,
                                                           (short)0, acc[0][0], false, false);
        acc[0][1] = __builtin_amdgcn_wmma_f32_16x16x32_f16(false, a0, false, b1,
                                                           (short)0, acc[0][1], false, false);
        acc[1][0] = __builtin_amdgcn_wmma_f32_16x16x32_f16(false, a1, false, b0,
                                                           (short)0, acc[1][0], false, false);
        acc[1][1] = __builtin_amdgcn_wmma_f32_16x16x32_f16(false, a1, false, b1,
                                                           (short)0, acc[1][1], false, false);
    }

    const float* biasB = bias ? bias + (long)(bz % aMod) * biasStride : nullptr;
    #pragma unroll
    for (int ti = 0; ti < 2; ++ti) {
        #pragma unroll
        for (int tj = 0; tj < 2; ++tj) {
            #pragma unroll
            for (int r = 0; r < 8; ++r) {
                int mm = m0 + ti * 16 + r + 8 * kh;
                int nn = n0 + tj * 16 + l15;
                float v = acc[ti][tj][r] * alpha;
                if (biasB) v += biasB[mm];
                if (Res)   v += resT ? Res[(long)bz * sR + (size_t)nn * ldr + mm]
                                     : Res[(long)bz * sR + (size_t)mm * ldr + nn];
                if (ACT == 1) v = v > 0.f ? v : 0.f;
                if (ACT == 2) v = (v > 20.f) ? v : log1pf(__expf(v));
                if (CT) Cb[(size_t)nn * ldc + mm] = v;
                else    Cb[(size_t)mm * ldc + nn] = v;
            }
        }
    }
}

// ---------------------------------------------------------------------------
// LayerNorm over C=512 for each (b,l) row.  Input indexed b*L*C + l*ls + c*cs.
// Output (b,l,c) contiguous.
// ---------------------------------------------------------------------------
__global__ __launch_bounds__(128)
void ln_rows(const float* __restrict__ X, int ls, int cs,
             const float* __restrict__ g, const float* __restrict__ bb,
             float* __restrict__ Y)
{
    __shared__ float red[128];
    const int tid = threadIdx.x;
    const int row = blockIdx.x;            // b*L + l
    const int b = row >> 10, l = row & 1023;
    const float* xb = X + (long)b * (LL * CC);
    float xv[4];
    #pragma unroll
    for (int j = 0; j < 4; ++j) {
        int c = tid + 128 * j;
        xv[j] = xb[(long)l * ls + (long)c * cs];
    }
    float s = xv[0] + xv[1] + xv[2] + xv[3];
    red[tid] = s; __syncthreads();
    for (int t = 64; t > 0; t >>= 1) { if (tid < t) red[tid] += red[tid + t]; __syncthreads(); }
    float mean = red[0] / 512.f; __syncthreads();
    float vs = 0.f;
    #pragma unroll
    for (int j = 0; j < 4; ++j) { float d = xv[j] - mean; vs += d * d; }
    red[tid] = vs; __syncthreads();
    for (int t = 64; t > 0; t >>= 1) { if (tid < t) red[tid] += red[tid + t]; __syncthreads(); }
    float rstd = rsqrtf(red[0] / 512.f + 1e-5f);
    #pragma unroll
    for (int j = 0; j < 4; ++j) {
        int c = tid + 128 * j;
        Y[(long)row * 512 + c] = (xv[j] - mean) * rstd * g[c] + bb[c];
    }
}

// ---------------------------------------------------------------------------
// Depthwise 3x3 SAME conv + bias + SiLU.  xm = xz rows [0,D).  (b,d,h,w)
// ---------------------------------------------------------------------------
__global__ __launch_bounds__(256)
void dwconv_silu(const float* __restrict__ xz, const float* __restrict__ dww,
                 const float* __restrict__ dwb, float* __restrict__ out)
{
    long i = (long)blockIdx.x * 256 + threadIdx.x;
    if (i >= (long)BB * DD * LL) return;
    int l = (int)(i & 1023), d = (int)((i >> 10) & 1023), b = (int)(i >> 20);
    int h = l >> 5, w = l & 31;
    const float* in = xz + ((long)b * 2048 + d) * LL;
    float acc = dwb[d];
    #pragma unroll
    for (int dy = 0; dy < 3; ++dy) {
        int hh = h + dy - 1;
        if (hh < 0 || hh > 31) continue;
        #pragma unroll
        for (int dx = 0; dx < 3; ++dx) {
            int ww = w + dx - 1;
            if (ww < 0 || ww > 31) continue;
            acc += dww[d * 9 + dy * 3 + dx] * in[hh * 32 + ww];
        }
    }
    out[i] = acc / (1.f + __expf(-acc));          // silu
}

// ---------------------------------------------------------------------------
// Build 4-direction xs (b,k,d,l) from xconv (b,d,l).
// ---------------------------------------------------------------------------
__global__ __launch_bounds__(256)
void build_xs(const float* __restrict__ xcv, float* __restrict__ xs)
{
    long i = (long)blockIdx.x * 256 + threadIdx.x;
    if (i >= (long)BB * DD * LL) return;
    int l = (int)(i & 1023), d = (int)((i >> 10) & 1023), b = (int)(i >> 20);
    int h = l >> 5, w = l & 31;
    int lw = w * 32 + h;
    float v = xcv[i];
    long k0 = (((long)(b * 4 + 0) * DD) + d) * LL;
    long k1 = (((long)(b * 4 + 1) * DD) + d) * LL;
    long k2 = (((long)(b * 4 + 2) * DD) + d) * LL;
    long k3 = (((long)(b * 4 + 3) * DD) + d) * LL;
    xs[k0 + l]            = v;
    xs[k1 + lw]           = v;
    xs[k2 + (1023 - l)]   = v;
    xs[k3 + (1023 - lw)]  = v;
}

// ---------------------------------------------------------------------------
// Selective scan: thread per (b,k,d), state h[16] in registers, 1024 steps.
// B/C tiles staged through LDS.  ys written in-place over xs.
// ---------------------------------------------------------------------------
__global__ __launch_bounds__(128)
void scan_kernel(float* __restrict__ xs,
                 const float* __restrict__ delta,
                 const float* __restrict__ xdbl,    // (bz,64,L): rows 32..47 B, 48..63 C
                 const float* __restrict__ A_log,
                 const float* __restrict__ Ds)
{
    const int tid = threadIdx.x;
    const int b = blockIdx.z, k = blockIdx.y;
    const int d = blockIdx.x * 128 + tid;
    const int bz = b * 4 + k;
    const float* Bm  = xdbl + ((long)bz * 64 + 32) * LL;
    const float* Cmp = xdbl + ((long)bz * 64 + 48) * LL;
    const long base = ((long)bz * DD + d) * LL;

    float Arow[NST_];
    #pragma unroll
    for (int n = 0; n < NST_; ++n)
        Arow[n] = -__expf(A_log[((long)k * DD + d) * NST_ + n]);
    const float Dv = Ds[k * DD + d];
    float h[NST_];
    #pragma unroll
    for (int n = 0; n < NST_; ++n) h[n] = 0.f;

    __shared__ float Bs[NST_ * 64];
    __shared__ float Cs[NST_ * 64];
    for (int t0 = 0; t0 < LL; t0 += 64) {
        __syncthreads();
        for (int i = tid; i < NST_ * 64; i += 128) {
            int n = i >> 6, tt = i & 63;
            Bs[i] = Bm[(long)n * LL + t0 + tt];
            Cs[i] = Cmp[(long)n * LL + t0 + tt];
        }
        __syncthreads();
        for (int tt = 0; tt < 64; ++tt) {
            long idx = base + t0 + tt;
            float dt = delta[idx];
            float xv = xs[idx];
            float dx = dt * xv;
            float y = 0.f;
            #pragma unroll
            for (int n = 0; n < NST_; ++n) {
                float dA = __expf(dt * Arow[n]);
                h[n] = dA * h[n] + dx * Bs[(n << 6) + tt];
                y += h[n] * Cs[(n << 6) + tt];
            }
            xs[idx] = y + Dv * xv;                 // ys in-place
        }
    }
}

// ---------------------------------------------------------------------------
// Combine 4 directions -> ycomb (b,l,d) (d fastest for coalesced writes).
// ---------------------------------------------------------------------------
__global__ __launch_bounds__(256)
void combine_dirs(const float* __restrict__ ys, float* __restrict__ ycomb)
{
    long i = (long)blockIdx.x * 256 + threadIdx.x;   // (b,l,d), d fastest
    if (i >= (long)BB * LL * DD) return;
    int d = (int)(i & 1023), l = (int)((i >> 10) & 1023), b = (int)(i >> 20);
    int h = l >> 5, w = l & 31;
    int lw = w * 32 + h;
    long e0 = (((long)(b * 4 + 0) * DD) + d) * LL;
    long e1 = (((long)(b * 4 + 1) * DD) + d) * LL;
    long e2 = (((long)(b * 4 + 2) * DD) + d) * LL;
    long e3 = (((long)(b * 4 + 3) * DD) + d) * LL;
    ycomb[i] = ys[e0 + l] + ys[e2 + (1023 - l)] + ys[e1 + lw] + ys[e3 + (1023 - lw)];
}

// ---------------------------------------------------------------------------
// LN over D=1024 + gamma/beta + * silu(z).  ycomb (b,l,d) -> yact (b,l,d).
// ---------------------------------------------------------------------------
__global__ __launch_bounds__(256)
void ynorm_silu(const float* __restrict__ ycomb, const float* __restrict__ xz,
                const float* __restrict__ g, const float* __restrict__ bb,
                float* __restrict__ yact)
{
    __shared__ float red[256];
    const int tid = threadIdx.x;
    const int row = blockIdx.x;             // b*L + l
    const int b = row >> 10, l = row & 1023;
    const float* yr = ycomb + (long)row * DD;
    float xv[4];
    #pragma unroll
    for (int j = 0; j < 4; ++j) xv[j] = yr[tid + 256 * j];
    float s = xv[0] + xv[1] + xv[2] + xv[3];
    red[tid] = s; __syncthreads();
    for (int t = 128; t > 0; t >>= 1) { if (tid < t) red[tid] += red[tid + t]; __syncthreads(); }
    float mean = red[0] / 1024.f; __syncthreads();
    float vs = 0.f;
    #pragma unroll
    for (int j = 0; j < 4; ++j) { float dd2 = xv[j] - mean; vs += dd2 * dd2; }
    red[tid] = vs; __syncthreads();
    for (int t = 128; t > 0; t >>= 1) { if (tid < t) red[tid] += red[tid + t]; __syncthreads(); }
    float rstd = rsqrtf(red[0] / 1024.f + 1e-5f);
    #pragma unroll
    for (int j = 0; j < 4; ++j) {
        int d = tid + 256 * j;
        float z = xz[((long)b * 2048 + 1024 + d) * LL + l];
        float sz = z / (1.f + __expf(-z));
        yact[(long)row * DD + d] = ((xv[j] - mean) * rstd * g[d] + bb[d]) * sz;
    }
}

// ---------------------------------------------------------------------------
// Per-row softmax + rank mask (in-place on P, rows of 1024).
// rank[i] = #{j : s[j]>s[i] or (s[j]==s[i] and j<i)}  (stable descending sort)
// mask = pos ? (rank_clamped+1)^3 : 1 ;  P = softmax * mask
// ---------------------------------------------------------------------------
__global__ __launch_bounds__(256)
void attn_mask_kernel(float* __restrict__ P)
{
    __shared__ float srow[1024];
    __shared__ float red[256];
    const int tid = threadIdx.x;
    float* pr = P + (long)blockIdx.x * 1024;
    #pragma unroll
    for (int j = 0; j < 4; ++j) srow[tid + 256 * j] = pr[tid + 256 * j];
    __syncthreads();
    float mx = -3.4e38f;
    #pragma unroll
    for (int j = 0; j < 4; ++j) mx = fmaxf(mx, srow[tid + 256 * j]);
    red[tid] = mx; __syncthreads();
    for (int t = 128; t > 0; t >>= 1) { if (tid < t) red[tid] = fmaxf(red[tid], red[tid + t]); __syncthreads(); }
    mx = red[0]; __syncthreads();
    float se = 0.f, pc = 0.f;
    #pragma unroll
    for (int j = 0; j < 4; ++j) {
        float v = srow[tid + 256 * j];
        se += __expf(v - mx);
        pc += (v >= 0.f) ? 1.f : 0.f;
    }
    red[tid] = se; __syncthreads();
    for (int t = 128; t > 0; t >>= 1) { if (tid < t) red[tid] += red[tid + t]; __syncthreads(); }
    se = red[0]; __syncthreads();
    red[tid] = pc; __syncthreads();
    for (int t = 128; t > 0; t >>= 1) { if (tid < t) red[tid] += red[tid + t]; __syncthreads(); }
    pc = red[0]; __syncthreads();

    #pragma unroll
    for (int j = 0; j < 4; ++j) {
        int i = tid + 256 * j;
        float v = srow[i];
        float cnt = 0.f;
        for (int q = 0; q < 1024; ++q) {
            float u = srow[q];
            cnt += ((u > v) || (u == v && q < i)) ? 1.f : 0.f;
        }
        float sp = (cnt > pc - 1.f) ? 0.f : cnt;
        float posf = (v >= 0.f) ? 1.f : 0.f;
        float sp1 = sp + 1.f;
        float m = sp1 * sp1 * sp1 * posf + (1.f - posf);
        pr[i] = (__expf(v - mx) / se) * m;
    }
}

__global__ __launch_bounds__(256)
void final_add(const float* __restrict__ a, const float* __restrict__ b,
               float* __restrict__ o, long n)
{
    long i = (long)blockIdx.x * 256 + threadIdx.x;
    if (i < n) o[i] = a[i] + b[i];
}

// ---------------------------------------------------------------------------
extern "C" void kernel_launch(void* const* d_in, const int* in_sizes, int n_in,
                              void* d_out, int out_size, void* d_ws, size_t ws_size,
                              hipStream_t stream)
{
    const float* x      = (const float*)d_in[0];
    const float* conv_w = (const float*)d_in[1];
    const float* conv_b = (const float*)d_in[2];
    const float* qkv_w  = (const float*)d_in[3];
    const float* qkv_b  = (const float*)d_in[4];
    const float* ln1_g  = (const float*)d_in[5];
    const float* ln1_b  = (const float*)d_in[6];
    const float* in_w   = (const float*)d_in[7];
    const float* dw_w   = (const float*)d_in[8];
    const float* dw_b   = (const float*)d_in[9];
    const float* xp_w   = (const float*)d_in[10];
    const float* dtp_w  = (const float*)d_in[11];
    const float* dtp_b  = (const float*)d_in[12];
    const float* A_log  = (const float*)d_in[13];
    const float* Ds     = (const float*)d_in[14];
    const float* on_g   = (const float*)d_in[15];
    const float* on_b   = (const float*)d_in[16];
    const float* op_w   = (const float*)d_in[17];
    float* out = (float*)d_out;

    // ---- workspace carving (floats) ----
    float* w = (float*)d_ws;
    size_t o = 0;
    auto alloc = [&](size_t n) { float* p = w + o; o += n; return p; };
    float* xc   = alloc(1048576);  // (b,c,l)
    float* xln  = alloc(1048576);  // (b,l,c)
    float* xz   = alloc(4194304);  // (b,2D,l)
    float* xcv  = alloc(2097152);  // (b,d,l) dwconv  /  ycomb (b,l,d)
    float* xs   = alloc(8388608);  // (b,k,d,l) ; ys in-place ; attn temps alias
    float* xdl  = alloc(524288);   // (bz,64,l)
    float* dlt  = alloc(8388608);  // (bz,d,l)
    float* yact = alloc(2097152);  // (b,l,d)
    float* x1   = alloc(1048576);  // (b,c,l)
    float* aout = alloc(1048576);  // (b,l,c)
    float* x2   = alloc(1048576);  // (b,c,l)
    // attention temporaries aliased into xs (dead during attention phase)
    float* qb = xs;                // (b,l,c)
    float* kb = xs + 1048576;      // (b,c,l)
    float* vb = xs + 2097152;      // (b,m,c)
    float* t1 = xs + 3145728;      // (b,c,l)
    float* pb = xs + 4194304;      // (b,l,m)  2M floats

    const long sBL = 524288;   // C*L per-batch
    const long sDL = 1048576;  // D*L per-batch

    // ---- xc = relu(conv_w @ x + conv_b) -------------------------------------
    gemm_wmma<false,false,1><<<dim3(8,16,2),128,0,stream>>>(
        conv_w,512,0,1, x,1024,sBL, xc,1024,sBL,
        conv_b,0, nullptr,0,0,0, 1.f, 512,1024,512);

    // ---- VSS block ----------------------------------------------------------
    auto run_vss = [&](const float* in, int ls, int cs,
                       const float* resid, int resT, int ldr, float* outp) {
        ln_rows<<<2048,128,0,stream>>>(in, ls, cs, ln1_g, ln1_b, xln);
        // in_proj: xz[o][l] = sum_c in_w[o][c]*xln[l][c]
        gemm_wmma<true,false,0><<<dim3(8,64,2),128,0,stream>>>(
            in_w,512,0,1, xln,512,sBL, xz,1024,(long)2097152,
            nullptr,0, nullptr,0,0,0, 1.f, 2048,1024,512);
        dwconv_silu<<<8192,256,0,stream>>>(xz, dw_w, dw_b, xcv);
        build_xs<<<8192,256,0,stream>>>(xcv, xs);
        // x_dbl[bz][c][l] = sum_d xp_w[k][c][d]*xs[bz][d][l]
        gemm_wmma<false,false,0><<<dim3(8,2,8),128,0,stream>>>(
            xp_w,1024,65536,4, xs,1024,sDL, xdl,1024,(long)65536,
            nullptr,0, nullptr,0,0,0, 1.f, 64,1024,1024);
        // delta = softplus(dtp_w @ dt + dtp_b)
        gemm_wmma<false,false,2><<<dim3(8,32,8),128,0,stream>>>(
            dtp_w,32,32768,4, xdl,1024,(long)65536, dlt,1024,sDL,
            dtp_b,1024, nullptr,0,0,0, 1.f, 1024,1024,32);
        scan_kernel<<<dim3(8,4,2),128,0,stream>>>(xs, dlt, xdl, A_log, Ds);
        combine_dirs<<<8192,256,0,stream>>>(xs, xcv);
        ynorm_silu<<<2048,256,0,stream>>>(xcv, xz, on_g, on_b, yact);
        // out_proj + residual
        gemm_wmma<true,false,0><<<dim3(8,16,2),128,0,stream>>>(
            op_w,1024,0,1, yact,1024,sDL, outp,1024,sBL,
            nullptr,0, resid,ldr,sBL,resT, 1.f, 512,1024,1024);
    };

    // VSS #1: input x0 = xc^T (layout (b,c,l): ls=1, cs=1024), residual xc
    run_vss(xc, 1, 1024, xc, 0, 1024, x1);

    // ---- attention q/k/v (two chained 1x1 convs each) -----------------------
    for (int i = 0; i < 3; ++i) {
        const float* w0 = qkv_w + (size_t)i * 2 * 262144;
        const float* w1 = w0 + 262144;
        const float* b0 = qkv_b + (size_t)i * 2 * 512;
        const float* b1 = b0 + 512;
        gemm_wmma<false,false,0><<<dim3(8,16,2),128,0,stream>>>(
            w0,512,0,1, x1,1024,sBL, t1,1024,sBL,
            b0,0, nullptr,0,0,0, 1.f, 512,1024,512);
        if (i == 1) {
            gemm_wmma<false,false,0><<<dim3(8,16,2),128,0,stream>>>(
                w1,512,0,1, t1,1024,sBL, kb,1024,sBL,
                b1,0, nullptr,0,0,0, 1.f, 512,1024,512);
        } else {
            gemm_wmma<false,true,0><<<dim3(8,16,2),128,0,stream>>>(
                w1,512,0,1, t1,1024,sBL, (i==0?qb:vb),512,sBL,
                b1,0, nullptr,0,0,0, 1.f, 512,1024,512);
        }
    }
    // bmm = (q @ k) / sqrt(C)
    gemm_wmma<false,false,0><<<dim3(8,32,2),128,0,stream>>>(
        qb,512,sBL,2, kb,1024,sBL, pb,1024,sDL,
        nullptr,0, nullptr,0,0,0, 0.04419417382415922f, 1024,1024,512);
    attn_mask_kernel<<<2048,256,0,stream>>>(pb);
    // attnout = P @ v   -> (b,l,c)
    gemm_wmma<false,false,0><<<dim3(4,32,2),128,0,stream>>>(
        pb,1024,sDL,2, vb,512,sBL, aout,512,sBL,
        nullptr,0, nullptr,0,0,0, 1.f, 1024,512,1024);

    // VSS #2: input attnout (b,l,c): ls=512, cs=1; residual attnout (transposed)
    run_vss(aout, 512, 1, aout, 1, 512, x2);

    // out = x2 + xc
    final_add<<<4096,256,0,stream>>>(x2, xc, out, (long)1048576);
}
